// RNNJANSEN_22325240005143
// MI455X (gfx1250) — compile-verified, matrix operand order checked by reading, not run on hardware
//
#include <hip/hip_runtime.h>
#include <hip/hip_bf16.h>

typedef float     v2f  __attribute__((ext_vector_type(2)));
typedef float     v8f  __attribute__((ext_vector_type(8)));
typedef _Float16  v16h __attribute__((ext_vector_type(16)));

#ifndef __has_builtin
#define __has_builtin(x) 0
#endif
#if __has_builtin(__builtin_amdgcn_wmma_f32_16x16x4_f32)
#define EEG_WMMA_F32 1
#else
#define EEG_WMMA_F32 0
#endif

namespace {
constexpr int N = 512, H = 40, B = 8, D = 384, O = 62;
constexpr float DT = 1e-4f, BOUND = 500.0f;
constexpr float A_ = 3.25f, a_ = 100.0f, Bp_ = 22.0f, b_ = 50.0f, g_ = 100.0f;
constexpr float c1 = 135.0f, c2 = 135.0f * 0.8f, c3 = 135.0f * 0.25f, c4 = 135.0f * 0.25f;
constexpr float STD_IN = 100.0f, VMAX = 5.0f, V0 = 6.0f, R_ = 0.56f, MU = 0.5f, KKI = 5.0f;

// d_out layout (float element offsets), reference return order:
// (current_state, eegs, Es, Is, Ms, Evs, Ivs, Mvs, hEf)
constexpr int O_STATE = 0;                  // (N,6)
constexpr int O_EEG   = N * 6;              // (O,B)
constexpr int O_ES    = O_EEG + O * B;      // (N,B)
constexpr int O_IS    = O_ES  + N * B;
constexpr int O_MS    = O_IS  + N * B;
constexpr int O_EVS   = O_MS  + N * B;
constexpr int O_IVS   = O_EVS + N * B;
constexpr int O_MVS   = O_IVS + N * B;
constexpr int O_HE    = O_MVS + N * B;      // (N,D)
}

__device__ __forceinline__ float sigm(float x) {
    return VMAX / (1.0f + expf(R_ * (V0 - x)));
}

// ---------------------------------------------------------------------------
// prep_ws: ws[i,j] = log1p(0.5*(exp(wbb[i,j])*sc[i,j] + exp(wbb[j,i])*sc[j,i]))
// (symmetric). Also per-row sum and per-row sum-of-squares (deterministic).
// One block per row i, 256 threads.
// ---------------------------------------------------------------------------
__global__ void prep_ws(const float* __restrict__ wbb, const float* __restrict__ sc,
                        float* __restrict__ ws, float* __restrict__ rowsumWS,
                        float* __restrict__ sqpart) {
    const int i = blockIdx.x;
    const int t = threadIdx.x;
    float s = 0.0f, q = 0.0f;
    for (int j = t; j < N; j += 256) {
        const float wij = expf(wbb[i * N + j]) * sc[i * N + j];
        const float wji = expf(wbb[j * N + i]) * sc[j * N + i];
        const float v   = log1pf(0.5f * (wij + wji));
        ws[i * N + j] = v;
        s += v;
        q += v * v;
    }
    __shared__ float ss[256];
    __shared__ float sq[256];
    ss[t] = s; sq[t] = q;
    __syncthreads();
    for (int o = 128; o > 0; o >>= 1) {
        if (t < o) { ss[t] += ss[t + o]; sq[t] += sq[t + o]; }
        __syncthreads();
    }
    if (t == 0) { rowsumWS[i] = ss[0]; sqpart[i] = sq[0]; }
}

// ---------------------------------------------------------------------------
// prep_norm: Frobenius norm of ws from the 512 per-row partials.
// ---------------------------------------------------------------------------
__global__ void prep_norm(const float* __restrict__ sqpart, float* __restrict__ normv) {
    __shared__ float s[512];
    const int t = threadIdx.x;
    s[t] = sqpart[t];
    __syncthreads();
    for (int o = 256; o > 0; o >>= 1) {
        if (t < o) s[t] += s[t + o];
        __syncthreads();
    }
    if (t == 0) normv[0] = sqrtf(s[0]);
}

// ---------------------------------------------------------------------------
// jansen_sim: persistent single-workgroup time loop. One thread = one node.
// hE lives in-place in the d_out hEf region. Final EEG projection uses WMMA.
// ---------------------------------------------------------------------------
__global__ __launch_bounds__(512, 1) void jansen_sim(
    const float* __restrict__ in_u, const float* __restrict__ in_nz,
    const float* __restrict__ hx, const int* __restrict__ delays,
    const float* __restrict__ lm, const float* __restrict__ wsym,
    const float* __restrict__ rowsumWS, const float* __restrict__ normv,
    float* __restrict__ out)
{
    const int n = threadIdx.x;
    __shared__ float sX[N * 16];   // X[k][col]: B-operand for EEG GEMM, cols 8..15 = 0
    #pragma unroll
    for (int c0 = 0; c0 < 16; ++c0) sX[n * 16 + c0] = 0.0f;

    const float inv = 1.0f / normv[0];
    const float rs  = rowsumWS[n] * inv;            // rowsum of normalized w_n
    float M  = hx[n * 6 + 0], E  = hx[n * 6 + 1], I  = hx[n * 6 + 2];
    float Mv = hx[n * 6 + 3], Ev = hx[n * 6 + 4], Iv = hx[n * 6 + 5];
    float* hE = out + O_HE;

    __builtin_prefetch(&wsym[n], 0, 1);             // global_prefetch_b8
    __syncthreads();

    for (int b = 0; b < B; ++b) {
        for (int h = 0; h < H; ++h) {
            // LEd[n]*norm = sum_j ws[j,n] * hE[j, delays[j,n]]  (ws symmetric)
            float acc = 0.0f;
            #pragma unroll 4
            for (int j = 0; j < N; ++j) {
                const int d = delays[j * N + n];
                acc = fmaf(wsym[j * N + n], hE[j * D + d], acc);
            }
            __syncthreads();    // all gathers done before anyone writes hE[:,0]

            const float u   = KKI * in_u[(n * H + h) * B + b];
            const float nE  = STD_IN * in_nz[(n * H + h) * B + b];
            const float EmI = E - I;
            const float rM  = sigm(EmI);
            const float rE  = c2 * sigm(c1 * M);
            const float rI  = c4 * sigm(c3 * M);
            const float coupled  = g_ * (acc * inv - rs * EmI);
            const float coupling = BOUND * tanhf(coupled * (1.0f / BOUND));

            const float Mn  = M + DT * Mv;
            const float En  = E + DT * Ev;
            const float In  = I + DT * Iv;
            const float Mvn = Mv + DT * (A_ * a_ * rM - 2.0f * a_ * Mv - a_ * a_ * M);
            const float Evn = Ev + DT * (A_ * a_ * (rE + coupling + u + nE + MU)
                                         - 2.0f * a_ * Ev - a_ * a_ * E);
            const float Ivn = Iv + DT * (Bp_ * b_ * rI - 2.0f * b_ * Iv - b_ * b_ * I);
            M = Mn; E = En; I = In; Mv = Mvn; Ev = Evn; Iv = Ivn;

            hE[n * D] = E - I;                       // hE[:,0] = En - In
            __threadfence();
            __syncthreads();
        }

        // hE = concat([E-I, hE[:, :-1]]): shift own row right by one
        const float emi = E - I;
        for (int d = D - 1; d >= 1; --d) hE[n * D + d] = hE[n * D + d - 1];
        hE[n * D] = emi;

        // per-batch-step outputs
        out[O_ES  + n * B + b] = E;
        out[O_IS  + n * B + b] = I;
        out[O_MS  + n * B + b] = M;
        out[O_EVS + n * B + b] = Ev;
        out[O_IVS + n * B + b] = Iv;
        out[O_MVS + n * B + b] = Mv;
        sX[n * 16 + b] = emi;                        // EEG GEMM B-operand column

        __threadfence();
        __syncthreads();
    }

    out[O_STATE + n * 6 + 0] = M;
    out[O_STATE + n * 6 + 1] = E;
    out[O_STATE + n * 6 + 2] = I;
    out[O_STATE + n * 6 + 3] = Mv;
    out[O_STATE + n * 6 + 4] = Ev;
    out[O_STATE + n * 6 + 5] = Iv;
    __syncthreads();
#if __has_builtin(__builtin_amdgcn_s_wait_tensorcnt)
    __builtin_amdgcn_s_wait_tensorcnt(0);
#endif

    // ---- EEG: (62x512) @ (512x8) via WMMA, waves 0..3 handle 16-row tiles ----
    if (threadIdx.x < 128) {                         // wave-uniform predicate
        const int wave = threadIdx.x >> 5;
        const int lane = threadIdx.x & 31;
        const int hi   = lane >> 4;                  // lane half (0/1)
        const int l16  = lane & 15;
        const int m    = wave * 16 + l16;            // A-matrix row this lane owns
        // Rows 62/63 of the padded 64-row A are zero: clamp the address (always
        // in-bounds -> unconditional loads, no EXEC predication) and zero via a
        // VALU mask multiply instead of branches.
        const int   mc    = (m < O) ? m : 0;
        const float amask = (m < O) ? 1.0f : 0.0f;
        const float* __restrict__ lmrow = lm + (size_t)mc * N;
        v8f c = {0.0f, 0.0f, 0.0f, 0.0f, 0.0f, 0.0f, 0.0f, 0.0f};

#if EEG_WMMA_F32
        for (int k0 = 0; k0 < N; k0 += 4) {
            v2f a, bf;
            const int ka = k0 + 2 * hi;              // A 16x4: lanes0-15 K={0,1}, lanes16-31 K={2,3}
            a.x = lmrow[ka]     * amask;
            a.y = lmrow[ka + 1] * amask;
            bf.x = sX[(k0 + hi)     * 16 + l16];     // B 4x16: VGPR0 rows K={0,1}, VGPR1 rows K={2,3}
            bf.y = sX[(k0 + 2 + hi) * 16 + l16];
            c = __builtin_amdgcn_wmma_f32_16x16x4_f32(false, a, false, bf,
                                                      (short)0, c, false, false);
        }
#else
        for (int k0 = 0; k0 < N; k0 += 32) {
            v16h a, bf;
            const int kb = k0 + hi * 8;              // 16-bit A 16x32 layout (ISA 7.12.2)
            #pragma unroll
            for (int q = 0; q < 8; ++q) {
                const int kA0 = kb + q;
                const int kA1 = kb + 16 + q;
                a[q]     = (_Float16)(lmrow[kA0] * amask);
                a[q + 8] = (_Float16)(lmrow[kA1] * amask);
                bf[q]     = (_Float16)sX[kA0 * 16 + l16];
                bf[q + 8] = (_Float16)sX[kA1 * 16 + l16];
            }
            c = __builtin_amdgcn_wmma_f32_16x16x32_f16(false, a, false, bf,
                                                       (short)0, c, false, false);
        }
#endif
        #pragma unroll
        for (int r = 0; r < 8; ++r) {                // D: VGPR r -> M = r (lo) / r+8 (hi)
            const int mm = wave * 16 + r + hi * 8;
            if (mm < O && l16 < B) out[O_EEG + mm * B + l16] = 0.0005f * c[r];
        }
    }
}

// ---------------------------------------------------------------------------
extern "C" void kernel_launch(void* const* d_in, const int* in_sizes, int n_in,
                              void* d_out, int out_size, void* d_ws, size_t ws_size,
                              hipStream_t stream) {
    (void)in_sizes; (void)n_in; (void)out_size; (void)ws_size;
    const float* in_u  = (const float*)d_in[0];   // input   (N,H,B)
    const float* in_nz = (const float*)d_in[1];   // noise_in(N,H,B,1)
    const float* hx    = (const float*)d_in[3];   // hx      (N,6)
    const float* hE0   = (const float*)d_in[4];   // hE      (N,D)
    const float* wbb   = (const float*)d_in[5];   // w_bb    (N,N)
    const float* sc    = (const float*)d_in[6];   // sc      (N,N)
    const float* lm    = (const float*)d_in[7];   // lm      (O,N)
    const int*   dly   = (const int*)d_in[8];     // delays  (N,N)
    float* out = (float*)d_out;

    float* wsbuf  = (float*)d_ws;                 // N*N
    float* rowsum = wsbuf + N * N;                // N
    float* sqpart = rowsum + N;                   // N
    float* normv  = sqpart + N;                   // 1

    // hE works in-place inside d_out's hEf region (it IS the hEf output)
    hipMemcpyAsync(out + O_HE, hE0, (size_t)N * D * sizeof(float),
                   hipMemcpyDeviceToDevice, stream);
    prep_ws  <<<N, 256, 0, stream>>>(wbb, sc, wsbuf, rowsum, sqpart);
    prep_norm<<<1, 512, 0, stream>>>(sqpart, normv);
    jansen_sim<<<1, 512, 0, stream>>>(in_u, in_nz, hx, dly, lm, wsbuf, rowsum, normv, out);
}
